// CoAttention_44074954392267
// MI455X (gfx1250) — compile-verified, matrix operand order checked by reading, not run on hardware
//
#include <hip/hip_runtime.h>
#include <hip/hip_fp16.h>

// ---------------- problem constants (match reference) ----------------
#define NN     32768
#define FIN    64
#define HID    256
#define NB     1024
#define EIN    262144
#define EOUT   524288
#define NCYC   2
#define SLOPE  0.01f

typedef _Float16 h16;
typedef __attribute__((ext_vector_type(16))) _Float16 v16h;
typedef __attribute__((ext_vector_type(8)))  float    v8f;

// ---------------- elementwise / scatter kernels ----------------

__global__ void k_zero_f32(float* __restrict__ p, long long n) {
  long long i = (long long)blockIdx.x * blockDim.x + threadIdx.x;
  long long s = (long long)gridDim.x * blockDim.x;
  for (; i < n; i += s) p[i] = 0.0f;
}

__global__ void k_cvt_f16(const float* __restrict__ in, h16* __restrict__ out, long long n) {
  long long i = (long long)blockIdx.x * blockDim.x + threadIdx.x;
  long long s = (long long)gridDim.x * blockDim.x;
  for (; i < n; i += s) out[i] = (h16)in[i];
}

// W: [Ktot, 256] row-major (rows koff..koff+K-1)  ->  Wt: [256, K] f16
__global__ void k_transpose_w(const float* __restrict__ W, h16* __restrict__ Wt,
                              int K, int koff) {
  long long n = (long long)HID * K;
  long long i = (long long)blockIdx.x * blockDim.x + threadIdx.x;
  long long s = (long long)gridDim.x * blockDim.x;
  for (; i < n; i += s) {
    int col = (int)(i / K);     // output feature n
    int k   = (int)(i % K);
    Wt[i] = (h16)W[(long long)(k + koff) * HID + col];
  }
}

// agg[dst[e]*F2 + f] += feat[src[e]*F2 + f]   (packed 2xf16 atomic add)
__global__ void k_scatter_add_pk(const __half2* __restrict__ feat,
                                 const int* __restrict__ src, const int* __restrict__ dst,
                                 __half2* __restrict__ agg, long long E, int F2) {
  long long n = E * F2;
  long long i = (long long)blockIdx.x * blockDim.x + threadIdx.x;
  long long s = (long long)gridDim.x * blockDim.x;
  for (; i < n; i += s) {
    long long e = i / F2;
    int f = (int)(i % F2);
    atomicAdd(&agg[(long long)dst[e] * F2 + f], feat[(long long)src[e] * F2 + f]);
  }
}

__global__ void k_count(const int* __restrict__ dst, float* __restrict__ deg, long long E) {
  long long i = (long long)blockIdx.x * blockDim.x + threadIdx.x;
  long long s = (long long)gridDim.x * blockDim.x;
  for (; i < E; i += s) atomicAdd(&deg[dst[i]], 1.0f);
}

// in-place: agg[i] /= max(deg[row],1)
__global__ void k_div_h16(h16* __restrict__ agg, const float* __restrict__ deg,
                          long long n, int F) {
  long long i = (long long)blockIdx.x * blockDim.x + threadIdx.x;
  long long s = (long long)gridDim.x * blockDim.x;
  for (; i < n; i += s) {
    long long row = i / F;
    agg[i] = (h16)((float)agg[i] / fmaxf(deg[row], 1.0f));
  }
}

__global__ void k_pool_add(const h16* __restrict__ h, const int* __restrict__ batch,
                           float* __restrict__ p, long long n) {
  long long i = (long long)blockIdx.x * blockDim.x + threadIdx.x;
  long long s = (long long)gridDim.x * blockDim.x;
  for (; i < n; i += s) {
    long long node = i >> 8;           // /HID
    int f = (int)(i & (HID - 1));
    atomicAdd(&p[(long long)batch[node] * HID + f], (float)h[i]);
  }
}

__global__ void k_pool_count(const int* __restrict__ batch, float* __restrict__ cnt, int n) {
  int i = blockIdx.x * blockDim.x + threadIdx.x;
  int s = gridDim.x * blockDim.x;
  for (; i < n; i += s) atomicAdd(&cnt[batch[i]], 1.0f);
}

// logits[b] = sigmoid( sum_f p_i[b,f]/ci * Wr[f] + p_j[b,f]/cj * Wr[256+f] + br )
__global__ void k_final(const float* __restrict__ p_i, const float* __restrict__ cnt_i,
                        const float* __restrict__ p_j, const float* __restrict__ cnt_j,
                        const float* __restrict__ Wr, const float* __restrict__ br,
                        float* __restrict__ out) {
  __shared__ float red[HID];
  int b = blockIdx.x;
  int f = threadIdx.x;
  float ci = fmaxf(cnt_i[b], 1.0f);
  float cj = fmaxf(cnt_j[b], 1.0f);
  float v = (p_i[(long long)b * HID + f] / ci) * Wr[f] +
            (p_j[(long long)b * HID + f] / cj) * Wr[HID + f];
  red[f] = v;
  __syncthreads();
  for (int s = HID / 2; s > 0; s >>= 1) {
    if (f < s) red[f] += red[f + s];
    __syncthreads();
  }
  if (f == 0) out[b] = 1.0f / (1.0f + expf(-(red[0] + br[0])));
}

// ---------------- WMMA dual GEMM ----------------
// ISA 16-bit A-matrix 16x32 layout (wave32): lane holds row m=lane&15;
// VGPR v holds K pair kk + (v/4)*16 + (lane>>4)*8 + (v&3)*2.
// B fragment mirrors this with lane = output column; B operands are read
// from an LDS-staged strip of the pre-transposed [256,K] weights so the
// 8 consecutive halves per VGPR-quad become ds_load_b128s.
__device__ __forceinline__ v16h load_frag16(const h16* base, int ld,
                                            int row, int kk, int hsel) {
  v16h f;
  const h16* p = base + (size_t)row * ld + kk + hsel * 8;
#pragma unroll
  for (int v = 0; v < 8; ++v) {
    int off = (v >> 2) * 16 + (v & 3) * 2;
    f[2 * v]     = p[off];
    f[2 * v + 1] = p[off + 1];
  }
  return f;
}

// CDNA5 async global->LDS copy (tracked by ASYNCcnt, bypasses VGPRs).
__device__ __forceinline__ void async_b128_to_lds(unsigned lds_addr, const void* g) {
  asm volatile("global_load_async_to_lds_b128 %0, %1, off"
               :: "v"(lds_addr), "v"(g)
               : "memory");
}
__device__ __forceinline__ void wait_asynccnt0() {
  asm volatile("s_wait_asynccnt 0x0" ::: "memory");
}

// out[M,256] = act( A1[M,K] @ Wt1^T + A2[M,K] @ Wt2^T + bias )
// Wt1/Wt2: [256, K] f16 (transposed weights).
// Block: 256 threads = 8 waves; block tile = 128 rows x 64 cols.
// Both 64-col weight strips (<=32KB each) staged to LDS once per block.
__global__ __launch_bounds__(256)
void dual_gemm_wmma(const h16* __restrict__ A1, const h16* __restrict__ Wt1,
                    const h16* __restrict__ A2, const h16* __restrict__ Wt2,
                    const float* __restrict__ bias, h16* __restrict__ out,
                    int K, int lrelu) {
  __shared__ __align__(16) h16 sB[2][64 * HID];   // [weight][r*K + k], 64KB

  const int lane = threadIdx.x & 31;
  const int wave = threadIdx.x >> 5;
  const int row0 = blockIdx.x * 128 + wave * 16;
  const int col0 = blockIdx.y * 64;
  const int mrow = row0 + (lane & 15);
  const int hsel = lane >> 4;
  const int ncol = lane & 15;

  // Capture sB through a laundered COPY of its address so the async asm's
  // "memory" clobber is assumed to store into it -- but keep all reads on the
  // original addrspace(3) array expression so they lower to ds_load_b128.
  // (Generic LDS pointers carry the LDS byte address in their low 32 bits.)
  h16* sB_cap = &sB[0][0];
  asm volatile("" : "+v"(sB_cap));
  const unsigned sbase = (unsigned)(unsigned long long)sB_cap;

  // async staging of the contiguous 64-row weight strips (global -> LDS DMA)
  {
    const uint4* g1 = (const uint4*)(Wt1 + (size_t)col0 * K);
    const uint4* g2 = (const uint4*)(Wt2 + (size_t)col0 * K);
    int nvec = 64 * K / 8;                        // 8 halves per 16B chunk
    for (int t = threadIdx.x; t < nvec; t += blockDim.x) {
      async_b128_to_lds(sbase + t * 16, g1 + t);
      async_b128_to_lds(sbase + 64 * HID * 2 + t * 16, g2 + t);
    }
    wait_asynccnt0();
  }
  __syncthreads();

  const h16* baseA1 = A1 + (size_t)mrow * K;
  const h16* baseA2 = A2 + (size_t)mrow * K;
  __builtin_prefetch(baseA1, 0, 3);
  __builtin_prefetch(baseA2, 0, 3);

  v8f acc[4] = {};

  // software pipeline: A fragments double-buffered across K-steps.
  // 2-entry buffers + unroll-2 make the buffer index a compile-time constant
  // in each unrolled copy (no register rotation moves).
  v16h a1b[2], a2b[2];
  a1b[0] = load_frag16(A1, K, mrow, 0, hsel);
  a2b[0] = load_frag16(A2, K, mrow, 0, hsel);
#pragma unroll 2
  for (int kk = 0; kk < K; kk += 32) {
    const int cur = (kk >> 5) & 1;
    const int nxt = cur ^ 1;
    if (kk + 32 < K) {
      a1b[nxt] = load_frag16(A1, K, mrow, kk + 32, hsel);
      a2b[nxt] = load_frag16(A2, K, mrow, kk + 32, hsel);
    }
    // batch all 8 B fragments so the ds_loads issue as one clause
    v16h b1[4], b2[4];
#pragma unroll
    for (int c = 0; c < 4; ++c) {
      b1[c] = load_frag16(&sB[0][0], K, c * 16 + ncol, kk, hsel);
      b2[c] = load_frag16(&sB[1][0], K, c * 16 + ncol, kk, hsel);
    }
#pragma unroll
    for (int c = 0; c < 4; ++c) {
      acc[c] = __builtin_amdgcn_wmma_f32_16x16x32_f16(false, a1b[cur], false, b1[c],
                                                      (short)0, acc[c], false, false);
      acc[c] = __builtin_amdgcn_wmma_f32_16x16x32_f16(false, a2b[cur], false, b2[c],
                                                      (short)0, acc[c], false, false);
    }
  }

#pragma unroll
  for (int c = 0; c < 4; ++c) {
    int col = col0 + c * 16 + ncol;
    float bv = bias ? bias[col] : 0.0f;
#pragma unroll
    for (int r = 0; r < 8; ++r) {
      float v = acc[c][r] + bv;                   // D row = r + 8*hsel
      if (lrelu) v = (v >= 0.0f) ? v : SLOPE * v;
      out[(size_t)(row0 + r + 8 * hsel) * HID + col] = (h16)v;
    }
  }
}

// ---------------- host-side orchestration ----------------

extern "C" void kernel_launch(void* const* d_in, const int* in_sizes, int n_in,
                              void* d_out, int out_size, void* d_ws, size_t ws_size,
                              hipStream_t stream) {
  const float* x_i     = (const float*)d_in[0];
  const float* x_j     = (const float*)d_in[1];
  const int*   ei_in_i = (const int*)d_in[2];   // [2, EIN]: row0 src, row1 dst
  const int*   ei_in_j = (const int*)d_in[3];
  const int*   ei_ot_i = (const int*)d_in[4];   // [2, EOUT]
  const int*   ei_ot_j = (const int*)d_in[5];
  const int*   batch_i = (const int*)d_in[6];
  const int*   batch_j = (const int*)d_in[7];
  const float* We_s    = (const float*)d_in[8];
  const float* We_n    = (const float*)d_in[9];
  const float* Win_s   = (const float*)d_in[10];
  const float* Win_n   = (const float*)d_in[11];
  const float* Wot_s   = (const float*)d_in[12];
  const float* Wot_n   = (const float*)d_in[13];
  const float* W_u     = (const float*)d_in[14]; // [512,256]
  const float* b_u     = (const float*)d_in[15];
  const float* W_r     = (const float*)d_in[16]; // [512,1]
  const float* b_r     = (const float*)d_in[17];
  float* out = (float*)d_out;                    // [NB]

  // workspace carve-out
  char* w = (char*)d_ws;
  auto alloc = [&](size_t b) { char* p = w; w += (b + 255) & ~(size_t)255; return p; };
  h16* tWe_s  = (h16*)alloc((size_t)HID * FIN * 2);
  h16* tWe_n  = (h16*)alloc((size_t)HID * FIN * 2);
  h16* tWin_s = (h16*)alloc((size_t)HID * HID * 2);
  h16* tWin_n = (h16*)alloc((size_t)HID * HID * 2);
  h16* tWot_s = (h16*)alloc((size_t)HID * HID * 2);
  h16* tWot_n = (h16*)alloc((size_t)HID * HID * 2);
  h16* tWu_t  = (h16*)alloc((size_t)HID * HID * 2);  // W_u rows 0..255
  h16* tWu_b  = (h16*)alloc((size_t)HID * HID * 2);  // W_u rows 256..511
  h16* x_i_h  = (h16*)alloc((size_t)NN * FIN * 2);
  h16* x_j_h  = (h16*)alloc((size_t)NN * FIN * 2);
  h16* h_i    = (h16*)alloc((size_t)NN * HID * 2);
  h16* h_j    = (h16*)alloc((size_t)NN * HID * 2);
  h16* m_i    = (h16*)alloc((size_t)NN * HID * 2);
  h16* m_j    = (h16*)alloc((size_t)NN * HID * 2);
  h16* a_ij   = (h16*)alloc((size_t)NN * HID * 2);
  h16* a_ji   = (h16*)alloc((size_t)NN * HID * 2);
  h16*   agg_h = (h16*)alloc((size_t)NN * HID * 2);  // f16 mean-agg accumulator
  float* deg   = (float*)alloc((size_t)NN * 4);
  float* p_i   = (float*)alloc((size_t)NB * HID * 4);
  float* p_j   = (float*)alloc((size_t)NB * HID * 4);
  float* cnt_i = (float*)alloc((size_t)NB * 4);
  float* cnt_j = (float*)alloc((size_t)NB * 4);

  const dim3 GS(8192), BS(256);

  // GraphConv(aggr='mean'): out = act( dst@Ws + mean_agg(src)@Wn )
  auto conv = [&](const h16* src_feat, const h16* dst_feat,
                  const int* ei, long long E, int F,
                  const h16* Wst, const h16* Wnt,
                  h16* outbuf, int lrelu) {
    long long nAgg = (long long)NN * F;
    k_zero_f32<<<GS, BS, 0, stream>>>((float*)agg_h, nAgg / 2);   // f16 zeros
    k_zero_f32<<<64, BS, 0, stream>>>(deg, (long long)NN);
    k_scatter_add_pk<<<GS, BS, 0, stream>>>((const __half2*)src_feat, ei, ei + E,
                                            (__half2*)agg_h, E, F / 2);
    k_count<<<GS, BS, 0, stream>>>(ei + E, deg, E);
    k_div_h16<<<GS, BS, 0, stream>>>(agg_h, deg, nAgg, F);
    dual_gemm_wmma<<<dim3(NN / 128, HID / 64), 256, 0, stream>>>(
        dst_feat, Wst, agg_h, Wnt, nullptr, outbuf, F, lrelu);
  };

  // --- weight prep (f16, transposed to [HID, K]) ---
  k_transpose_w<<<GS, BS, 0, stream>>>(We_s,  tWe_s,  FIN, 0);
  k_transpose_w<<<GS, BS, 0, stream>>>(We_n,  tWe_n,  FIN, 0);
  k_transpose_w<<<GS, BS, 0, stream>>>(Win_s, tWin_s, HID, 0);
  k_transpose_w<<<GS, BS, 0, stream>>>(Win_n, tWin_n, HID, 0);
  k_transpose_w<<<GS, BS, 0, stream>>>(Wot_s, tWot_s, HID, 0);
  k_transpose_w<<<GS, BS, 0, stream>>>(Wot_n, tWot_n, HID, 0);
  k_transpose_w<<<GS, BS, 0, stream>>>(W_u,   tWu_t,  HID, 0);
  k_transpose_w<<<GS, BS, 0, stream>>>(W_u,   tWu_b,  HID, HID);
  k_cvt_f16<<<GS, BS, 0, stream>>>(x_i, x_i_h, (long long)NN * FIN);
  k_cvt_f16<<<GS, BS, 0, stream>>>(x_j, x_j_h, (long long)NN * FIN);

  // --- encoder (no activation) ---
  conv(x_i_h, x_i_h, ei_in_i, EIN, FIN, tWe_s, tWe_n, h_i, 0);
  conv(x_j_h, x_j_h, ei_in_j, EIN, FIN, tWe_s, tWe_n, h_j, 0);

  // --- message-passing cycles ---
  for (int cyc = 0; cyc < NCYC; ++cyc) {
    conv(h_i, h_i, ei_in_i, EIN,  HID, tWin_s, tWin_n, m_i,  1);
    conv(h_j, h_j, ei_in_j, EIN,  HID, tWin_s, tWin_n, m_j,  1);
    conv(h_j, h_i, ei_ot_j, EOUT, HID, tWot_s, tWot_n, a_ij, 1);  // src j -> dst i
    conv(h_i, h_j, ei_ot_i, EOUT, HID, tWot_s, tWot_n, a_ji, 1);  // src i -> dst j
    // h = lrelu( [m, a] @ W_u + b_u ) == lrelu( m@Wu_top + a@Wu_bot + b_u )
    dual_gemm_wmma<<<dim3(NN / 128, HID / 64), 256, 0, stream>>>(
        m_i, tWu_t, a_ij, tWu_b, b_u, h_i, HID, 1);
    dual_gemm_wmma<<<dim3(NN / 128, HID / 64), 256, 0, stream>>>(
        m_j, tWu_t, a_ji, tWu_b, b_u, h_j, HID, 1);
  }

  // --- global mean pool + head ---
  k_zero_f32<<<GS, BS, 0, stream>>>(p_i, (long long)NB * HID);
  k_zero_f32<<<GS, BS, 0, stream>>>(p_j, (long long)NB * HID);
  k_zero_f32<<<16, BS, 0, stream>>>(cnt_i, (long long)NB);
  k_zero_f32<<<16, BS, 0, stream>>>(cnt_j, (long long)NB);
  k_pool_add<<<GS, BS, 0, stream>>>(h_i, batch_i, p_i, (long long)NN * HID);
  k_pool_add<<<GS, BS, 0, stream>>>(h_j, batch_j, p_j, (long long)NN * HID);
  k_pool_count<<<128, BS, 0, stream>>>(batch_i, cnt_i, NN);
  k_pool_count<<<128, BS, 0, stream>>>(batch_j, cnt_j, NN);
  k_final<<<NB, HID, 0, stream>>>(p_i, cnt_i, p_j, cnt_j, W_r, b_r, out);
}
